// LabelPropagation_53876069761353
// MI455X (gfx1250) — matrix operand order repaired and checked by reading, not compiled, not run on hardware
//
#include <hip/hip_runtime.h>
#include <hip/hip_bf16.h>

typedef __attribute__((ext_vector_type(16))) _Float16 v16h;
typedef __attribute__((ext_vector_type(8)))  float    v8f;

// Problem constants
#define BB   8
#define NN   512          // IDX + M
#define IDXN 256
#define DF   768
#define NH   12
#define HD   64
#define PJ   256
#define HC   128
#define VOC  16384
#define TK   20

// ---------------------------------------------------------------------------
// Generic WMMA GEMM:  C[M,N] = scale * (A[M,K] @ B) + bias
//   A row-major (lda), B either K-major [K,N] (transB=0, ldb) or row-major
//   [N,K] (transB=1, ldb).  Two nested batch dims (B1 outer, B2 inner) with
//   independent strides.  One wave computes one 16x16 C tile via
//   v_wmma_f32_16x16x32_f16 (f32->f16 operand conversion, f32 accumulate).
//   Fragment layouts per CDNA5 ISA 7.12.2.
// ---------------------------------------------------------------------------
__global__ void wmma_gemm_f16(const float* __restrict__ A,
                              const float* __restrict__ B,
                              float* __restrict__ C,
                              const float* __restrict__ bias,
                              int M, int N, int K,
                              int lda, int ldb, int ldc,
                              long long aS1, long long aS2,
                              long long bS1, long long bS2,
                              long long cS1, long long cS2,
                              int B2, int transB, float scale)
{
  int z  = blockIdx.z;
  int b1 = z / B2, b2 = z % B2;
  const float* Ab = A + b1 * aS1 + b2 * aS2;
  const float* Bb = B + b1 * bS1 + b2 * bS2;
  float*       Cb = C + b1 * cS1 + b2 * cS2;

  int tileN = blockIdx.x * 16;
  int tileM = (blockIdx.y * blockDim.y + threadIdx.y) * 16;
  if (tileM >= M || tileN >= N) return;   // wave-uniform: EXEC stays all-1s

  int lane = threadIdx.x;                 // 0..31 (wave32)
  int lo = lane & 15, hi = lane >> 4;

  v8f acc = {};
  const float* arow = Ab + (long long)(tileM + lo) * lda;

  for (int k0 = 0; k0 < K; k0 += 32) {
    // prefetch next A chunk (global_prefetch_b8)
    if (k0 + 32 < K) __builtin_prefetch(arow + k0 + 32, 0, 0);

    v16h af, bf;
    // A fragment (16-bit A 16x32): lane holds row M=lo; K per VGPR v,half h:
    //   K = 16*(v>>2) + 2*(v&3) + 8*hi + h
#pragma unroll
    for (int v = 0; v < 8; ++v) {
      int ka = k0 + 16 * (v >> 2) + 2 * (v & 3) + 8 * hi;
      af[2 * v]     = (_Float16)arow[ka];
      af[2 * v + 1] = (_Float16)arow[ka + 1];
    }
    // B fragment (16-bit B 32x16): lane holds col N=lo; K = 16*(v>>2) + 2*(v&3) + 8*hi + h
    if (transB) {
      const float* brow = Bb + (long long)(tileN + lo) * ldb;
#pragma unroll
      for (int v = 0; v < 8; ++v) {
        int kb = k0 + 16 * (v >> 2) + 2 * (v & 3) + 8 * hi;
        bf[2 * v]     = (_Float16)brow[kb];
        bf[2 * v + 1] = (_Float16)brow[kb + 1];
      }
    } else {
#pragma unroll
      for (int v = 0; v < 8; ++v) {
        int kb = k0 + 16 * (v >> 2) + 2 * (v & 3) + 8 * hi;
        bf[2 * v]     = (_Float16)Bb[(long long)kb * ldb + tileN + lo];
        bf[2 * v + 1] = (_Float16)Bb[(long long)(kb + 1) * ldb + tileN + lo];
      }
    }
    acc = __builtin_amdgcn_wmma_f32_16x16x32_f16(
        false, af, false, bf, (short)0, acc, false, false);
  }

  float bv = bias ? bias[tileN + lo] : 0.0f;
  // C/D layout: VGPR r -> M = r + 8*hi, N = lo
#pragma unroll
  for (int r = 0; r < 8; ++r) {
    Cb[(long long)(tileM + r + 8 * hi) * ldc + tileN + lo] = acc[r] * scale + bv;
  }
}

static inline void launch_gemm(const float* A, const float* B, float* C,
                               const float* bias, int M, int N, int K,
                               int lda, int ldb, int ldc,
                               long long aS1, long long aS2,
                               long long bS1, long long bS2,
                               long long cS1, long long cS2,
                               int B1, int B2, int transB, float scale,
                               hipStream_t stream)
{
  dim3 blk(32, 4, 1);
  dim3 grd((unsigned)(N / 16), (unsigned)((M + 63) / 64), (unsigned)(B1 * B2));
  wmma_gemm_f16<<<grd, blk, 0, stream>>>(A, B, C, bias, M, N, K, lda, ldb, ldc,
                                         aS1, aS2, bS1, bS2, cS1, cS2,
                                         B2, transB, scale);
}

// ---------------------------------------------------------------------------
// Elementwise / reduction kernels
// ---------------------------------------------------------------------------
__global__ void build_emb(const float* __restrict__ ge,
                          const float* __restrict__ ue,
                          float* __restrict__ emb)
{
  long long idx = (long long)blockIdx.x * blockDim.x + threadIdx.x;
  if (idx >= (long long)BB * NN * DF) return;
  int c = (int)(idx % DF);
  int n = (int)((idx / DF) % NN);
  int b = (int)(idx / ((long long)NN * DF));
  float base = (n < IDXN) ? ge[((long long)b * IDXN + n) * DF + c]
                          : ue[((long long)b * IDXN + (n - IDXN)) * DF + c];
  int j = c >> 1;
  float ang = (float)n * expf(-(float)(2 * j) * (9.210340371976184f / (float)DF));
  float pe  = (c & 1) ? cosf(ang) : sinf(ang);
  emb[idx] = base + pe;
}

__global__ void softmax512(float* __restrict__ x)
{
  float* row = x + (long long)blockIdx.x * 512;
  int t = threadIdx.x;
  float a = row[t], b = row[t + 256];
  __shared__ float sm[256];
  sm[t] = fmaxf(a, b); __syncthreads();
  for (int s = 128; s > 0; s >>= 1) { if (t < s) sm[t] = fmaxf(sm[t], sm[t + s]); __syncthreads(); }
  float mx = sm[0]; __syncthreads();
  float e0 = expf(a - mx), e1 = expf(b - mx);
  sm[t] = e0 + e1; __syncthreads();
  for (int s = 128; s > 0; s >>= 1) { if (t < s) sm[t] += sm[t + s]; __syncthreads(); }
  float inv = 1.0f / sm[0];
  row[t] = e0 * inv; row[t + 256] = e1 * inv;
}

__global__ void residual_ln(float* __restrict__ emb, const float* __restrict__ mha,
                            const float* __restrict__ g, const float* __restrict__ bt)
{
  int row = blockIdx.x;            // 0 .. B*256-1
  int b = row >> 8, r = row & 255;
  float* x = emb + ((long long)b * NN + IDXN + r) * DF;
  const float* mo = mha + ((long long)b * IDXN + r) * DF;
  int t = threadIdx.x;
  float v0 = x[t] + mo[t];
  float v1 = x[t + 256] + mo[t + 256];
  float v2 = x[t + 512] + mo[t + 512];
  __shared__ float sm[256];
  sm[t] = v0 + v1 + v2; __syncthreads();
  for (int s = 128; s > 0; s >>= 1) { if (t < s) sm[t] += sm[t + s]; __syncthreads(); }
  float mean = sm[0] * (1.0f / DF); __syncthreads();
  float d0 = v0 - mean, d1 = v1 - mean, d2 = v2 - mean;
  sm[t] = d0 * d0 + d1 * d1 + d2 * d2; __syncthreads();
  for (int s = 128; s > 0; s >>= 1) { if (t < s) sm[t] += sm[t + s]; __syncthreads(); }
  float inv = rsqrtf(sm[0] * (1.0f / DF) + 1e-5f);
  x[t]       = d0 * inv * g[t]       + bt[t];
  x[t + 256] = d1 * inv * g[t + 256] + bt[t + 256];
  x[t + 512] = d2 * inv * g[t + 512] + bt[t + 512];
}

// conv1d k=3 pad=1 over xp[b, t, i] (i = channels PJ), out h[b,o,t], fused BN+ReLU
__global__ void conv1_bn_relu(const float* __restrict__ xp, const float* __restrict__ w,
                              const float* __restrict__ cb, const float* __restrict__ g,
                              const float* __restrict__ bb, float* __restrict__ out)
{
  long long idx = (long long)blockIdx.x * blockDim.x + threadIdx.x;
  if (idx >= (long long)BB * HC * NN) return;
  int t = (int)(idx % NN);
  int o = (int)((idx / NN) % HC);
  int b = (int)(idx / ((long long)HC * NN));
  float s = 0.f;
  for (int dt = 0; dt < 3; ++dt) {
    int tt = t + dt - 1;
    if (tt < 0 || tt >= NN) continue;
    const float* xr = xp + ((long long)b * NN + tt) * PJ;
    const float* wr = w + (long long)o * PJ * 3 + dt;
    for (int i = 0; i < PJ; ++i) s += wr[i * 3] * xr[i];
  }
  const float invbn = 0.99999500003750f;  // 1/sqrt(1+1e-5)
  s = (s + cb[o]) * invbn * g[o] + bb[o];
  out[idx] = fmaxf(s, 0.f);
}

__global__ void conv2_bn_relu(const float* __restrict__ h1, const float* __restrict__ w,
                              const float* __restrict__ cb, const float* __restrict__ g,
                              const float* __restrict__ bb, float* __restrict__ out)
{
  long long idx = (long long)blockIdx.x * blockDim.x + threadIdx.x;
  if (idx >= (long long)BB * HC * NN) return;
  int t = (int)(idx % NN);
  int o = (int)((idx / NN) % HC);
  int b = (int)(idx / ((long long)HC * NN));
  float s = 0.f;
  for (int dt = 0; dt < 3; ++dt) {
    int tt = t + dt - 1;
    if (tt < 0 || tt >= NN) continue;
    const float* hr = h1 + (long long)b * HC * NN + tt;   // h1[b,i,tt]
    const float* wr = w + (long long)o * HC * 3 + dt;
    for (int i = 0; i < HC; ++i) s += wr[i * 3] * hr[(long long)i * NN];
  }
  const float invbn = 0.99999500003750f;
  s = (s + cb[o]) * invbn * g[o] + bb[o];
  out[idx] = fmaxf(s, 0.f);
}

__global__ void sigma_k(const float* __restrict__ h2, const float* __restrict__ ow,
                        const float* __restrict__ ob, float* __restrict__ sigma)
{
  int idx = blockIdx.x * blockDim.x + threadIdx.x;
  if (idx >= BB * NN) return;
  int t = idx % NN, b = idx / NN;
  float s = ob[0];
  const float* hb = h2 + (long long)b * HC * NN + t;
  for (int o = 0; o < HC; ++o) s += ow[o] * hb[(long long)o * NN];
  sigma[idx] = s;
}

__global__ void e_k(const float* __restrict__ emb, const float* __restrict__ sigma,
                    float* __restrict__ e)
{
  long long idx = (long long)blockIdx.x * blockDim.x + threadIdx.x;
  if (idx >= (long long)BB * NN * DF) return;
  long long bn = idx / DF;
  e[idx] = emb[idx] / (sigma[bn] + 1e-6f);
}

__global__ void diag_k(const float* __restrict__ G, float* __restrict__ diag)
{
  int idx = blockIdx.x * blockDim.x + threadIdx.x;
  if (idx >= BB * NN) return;
  int n = idx & (NN - 1), b = idx >> 9;
  diag[idx] = G[(long long)b * NN * NN + (long long)n * NN + n];
}

__global__ void wexp_k(float* __restrict__ G, const float* __restrict__ diag)
{
  long long idx = (long long)blockIdx.x * blockDim.x + threadIdx.x;
  if (idx >= (long long)BB * NN * NN) return;
  int m = (int)(idx % NN);
  int n = (int)((idx / NN) % NN);
  int b = (int)(idx / ((long long)NN * NN));
  float d2 = fmaxf(diag[b * NN + n] + diag[b * NN + m] - 2.0f * G[idx], 0.0f);
  G[idx] = expf(-d2 * (0.5f / (float)DF));
}

__global__ void zero_bytes(unsigned int* __restrict__ p, long long n_u32)
{
  for (long long i = (long long)blockIdx.x * blockDim.x + threadIdx.x;
       i < n_u32; i += (long long)gridDim.x * blockDim.x) p[i] = 0u;
}

__global__ void topk_mask(const float* __restrict__ W, unsigned char* __restrict__ mask)
{
  int idx = blockIdx.x * blockDim.x + threadIdx.x;
  if (idx >= BB * NN) return;
  int b = idx >> 9, n = idx & (NN - 1);
  const float* row = W + ((long long)b * NN + n) * NN;
  unsigned char* mb = mask + (long long)b * NN * NN;
  unsigned int fl[NN / 32];
#pragma unroll
  for (int i = 0; i < NN / 32; ++i) fl[i] = 0u;
  for (int t = 0; t < TK; ++t) {
    float best = -1e30f; int bi = 0;
    for (int m = 0; m < NN; ++m) {
      if (fl[m >> 5] & (1u << (m & 31))) continue;
      float v = row[m];
      if (v > best) { best = v; bi = m; }
    }
    fl[bi >> 5] |= 1u << (bi & 31);
    mb[(long long)n * NN + bi] = 1;     // mask OR mask^T > 0
    mb[(long long)bi * NN + n] = 1;
  }
}

__global__ void apply_mask(float* __restrict__ W, const unsigned char* __restrict__ mask)
{
  long long idx = (long long)blockIdx.x * blockDim.x + threadIdx.x;
  if (idx >= (long long)BB * NN * NN) return;
  W[idx] *= (float)mask[idx];
}

__global__ void rowsum_dsi(const float* __restrict__ W, float* __restrict__ dsi)
{
  const float* row = W + (long long)blockIdx.x * NN;
  int t = threadIdx.x;
  __shared__ float sm[256];
  sm[t] = row[t] + row[t + 256]; __syncthreads();
  for (int s = 128; s > 0; s >>= 1) { if (t < s) sm[t] += sm[t + s]; __syncthreads(); }
  if (t == 0) dsi[blockIdx.x] = sqrtf(1.0f / (sm[0] + 1e-6f));
}

__global__ void build_aug(const float* __restrict__ W, const float* __restrict__ dsi,
                          const float* __restrict__ alpha, float* __restrict__ aug)
{
  long long idx = (long long)blockIdx.x * blockDim.x + threadIdx.x;
  if (idx >= (long long)BB * NN * NN) return;
  int m = (int)(idx % NN);
  int n = (int)((idx / NN) % NN);
  int b = (int)(idx / ((long long)NN * NN));
  float a = alpha[0];
  float v = ((n == m) ? (1.0f + 1e-6f) : 0.0f)
            - a * dsi[b * NN + n] * dsi[b * NN + m] * W[idx];
  float* Ab = aug + (long long)b * NN * 1024;
  Ab[(long long)n * 1024 + m] = v;
  Ab[(long long)n * 1024 + 512 + m] = (n == m) ? 1.0f : 0.0f;
}

// Gauss-Jordan with partial pivoting on augmented [512 x 1024]; one block/batch.
__global__ void __launch_bounds__(1024) gauss_jordan(float* __restrict__ aug)
{
  float* A = aug + (long long)blockIdx.x * NN * 1024;
  int t = threadIdx.x;
  __shared__ float rv[1024];
  __shared__ int   ri[1024];
  for (int k = 0; k < NN; ++k) {
    // partial pivot search over rows >= k
    float bv = -1.0f; int bi = k;
    for (int r = k + t; r < NN; r += 1024) {
      float v = fabsf(A[(long long)r * 1024 + k]);
      if (v > bv) { bv = v; bi = r; }
    }
    rv[t] = bv; ri[t] = bi; __syncthreads();
    for (int s = 512; s > 0; s >>= 1) {
      if (t < s && rv[t + s] > rv[t]) { rv[t] = rv[t + s]; ri[t] = ri[t + s]; }
      __syncthreads();
    }
    int piv = ri[0];
    // swap rows k <-> piv (thread t owns column t)
    if (piv != k) {
      float tmp = A[(long long)k * 1024 + t];
      A[(long long)k * 1024 + t]   = A[(long long)piv * 1024 + t];
      A[(long long)piv * 1024 + t] = tmp;
    }
    __syncthreads();
    float pivval = A[(long long)k * 1024 + k];
    __syncthreads();
    float inv = 1.0f / pivval;
    A[(long long)k * 1024 + t] *= inv;
    __syncthreads();
    // eliminate: 2 threads per row, 512 columns each
    int r = t >> 1, half = t & 1;
    float f = A[(long long)r * 1024 + k];
    __syncthreads();
    if (r != k && f != 0.0f) {
      const float* pk = A + (long long)k * 1024 + half * 512;
      float*       pr = A + (long long)r * 1024 + half * 512;
      for (int c = 0; c < 512; ++c) pr[c] -= f * pk[c];
    }
    __syncthreads();
  }
}

// out[b,n,labels[b,m]] += inv[b][n][m]   (prop @ one_hot == column scatter)
__global__ void scatter_out(const float* __restrict__ aug, const int* __restrict__ labels,
                            float* __restrict__ out)
{
  int blk = blockIdx.x;
  int b = blk / NN, m = blk % NN;
  int v = labels[b * NN + m];
  if (v < 0 || v >= VOC) return;       // label==VOCAB -> all-zero one-hot row
  const float* inv = aug + (long long)b * NN * 1024 + 512 + m;
  float* ob = out + ((long long)b * NN) * VOC + v;
  for (int n = threadIdx.x; n < NN; n += blockDim.x)
    atomicAdd(ob + (long long)n * VOC, inv[(long long)n * 1024]);
}

// ---------------------------------------------------------------------------
extern "C" void kernel_launch(void* const* d_in, const int* in_sizes, int n_in,
                              void* d_out, int out_size, void* d_ws, size_t ws_size,
                              hipStream_t stream)
{
  (void)in_sizes; (void)n_in; (void)ws_size;
  const float* ge   = (const float*)d_in[0];
  const float* ue   = (const float*)d_in[1];
  const float* hs   = (const float*)d_in[2];
  const int*   lab  = (const int*)  d_in[3];
  const float* miw  = (const float*)d_in[4];
  const float* mib  = (const float*)d_in[5];
  const float* mow  = (const float*)d_in[6];
  const float* mob  = (const float*)d_in[7];
  const float* lng  = (const float*)d_in[8];
  const float* lnb  = (const float*)d_in[9];
  const float* pw   = (const float*)d_in[10];
  const float* pb   = (const float*)d_in[11];
  const float* c1w  = (const float*)d_in[12];
  const float* c1b  = (const float*)d_in[13];
  const float* bn1g = (const float*)d_in[14];
  const float* bn1b = (const float*)d_in[15];
  const float* c2w  = (const float*)d_in[16];
  const float* c2b  = (const float*)d_in[17];
  const float* bn2g = (const float*)d_in[18];
  const float* bn2b = (const float*)d_in[19];
  const float* ow   = (const float*)d_in[20];
  const float* ob   = (const float*)d_in[21];
  const float* alp  = (const float*)d_in[22];
  float* out = (float*)d_out;

  // workspace carve-up (floats)
  float* ws = (float*)d_ws;
  size_t off = 0;
  auto alloc = [&](size_t n) { float* p = ws + off; off += n; return p; };
  float* emb    = alloc((size_t)BB * NN * DF);        // 3.15M
  float* Qb     = alloc((size_t)BB * IDXN * DF);      // 1.57M
  float* Kb     = alloc((size_t)BB * NN * DF);        // 3.15M
  float* Vb     = alloc((size_t)BB * NN * DF);        // 3.15M
  float* scores = alloc((size_t)BB * NH * IDXN * NN); // 12.6M (reused as e)
  float* Ob     = alloc((size_t)BB * IDXN * DF);
  float* mtmp   = alloc((size_t)BB * IDXN * DF);
  float* xp     = alloc((size_t)BB * NN * PJ);
  float* h1     = alloc((size_t)BB * HC * NN);
  float* h2     = alloc((size_t)BB * HC * NN);
  float* sigma  = alloc((size_t)BB * NN);
  float* diag   = alloc((size_t)BB * NN);
  float* dsi    = alloc((size_t)BB * NN);
  float* Wb     = alloc((size_t)BB * NN * NN);        // 2.1M
  float* aug    = alloc((size_t)BB * NN * 1024);      // 4.2M
  unsigned char* mask = (unsigned char*)(ws + off);   // 2 MB
  float* e = scores;                                  // alias (scores dead by then)

  const long long embN = (long long)BB * NN * DF;
  const long long WN   = (long long)BB * NN * NN;

  // 1) emb = concat(ge, ue) + PE
  build_emb<<<dim3((unsigned)((embN + 255) / 256)), dim3(256), 0, stream>>>(ge, ue, emb);

  // 2) MHA:  Q = u @ wq^T + bq   (u = emb[:,256:,:]),  K/V from hidden_states
  launch_gemm(emb + (size_t)IDXN * DF, miw, Qb, mib,
              IDXN, DF, DF, DF, DF, DF,
              (long long)NN * DF, 0, 0, 0, (long long)IDXN * DF, 0,
              BB, 1, 1, 1.0f, stream);
  launch_gemm(hs, miw + (size_t)DF * DF, Kb, mib + DF,
              NN, DF, DF, DF, DF, DF,
              (long long)NN * DF, 0, 0, 0, (long long)NN * DF, 0,
              BB, 1, 1, 1.0f, stream);
  launch_gemm(hs, miw + (size_t)2 * DF * DF, Vb, mib + 2 * DF,
              NN, DF, DF, DF, DF, DF,
              (long long)NN * DF, 0, 0, 0, (long long)NN * DF, 0,
              BB, 1, 1, 1.0f, stream);

  // scores[b,h] = (Q_h @ K_h^T) / sqrt(64)
  launch_gemm(Qb, Kb, scores, nullptr,
              IDXN, NN, HD, DF, DF, NN,
              (long long)IDXN * DF, HD, (long long)NN * DF, HD,
              (long long)NH * IDXN * NN, (long long)IDXN * NN,
              BB, NH, 1, 0.125f, stream);
  softmax512<<<dim3(BB * NH * IDXN), dim3(256), 0, stream>>>(scores);

  // O[b,:,h*64:(h+1)*64] = attn @ V_h  (heads written interleaved -> concat)
  launch_gemm(scores, Vb, Ob, nullptr,
              IDXN, HD, NN, NN, DF, DF,
              (long long)NH * IDXN * NN, (long long)IDXN * NN,
              (long long)NN * DF, HD,
              (long long)IDXN * DF, HD,
              BB, NH, 0, 1.0f, stream);

  // out projection
  launch_gemm(Ob, mow, mtmp, mob,
              IDXN, DF, DF, DF, DF, DF,
              (long long)IDXN * DF, 0, 0, 0, (long long)IDXN * DF, 0,
              BB, 1, 1, 1.0f, stream);

  // 3) u = LN(u + mha) written back into emb[:,256:,:]
  residual_ln<<<dim3(BB * IDXN), dim3(256), 0, stream>>>(emb, mtmp, lng, lnb);

  // 4) xp = emb @ proj_w^T + proj_b
  launch_gemm(emb, pw, xp, pb,
              NN, PJ, DF, DF, DF, PJ,
              (long long)NN * DF, 0, 0, 0, (long long)NN * PJ, 0,
              BB, 1, 1, 1.0f, stream);

  // 5) convs + sigma
  long long hN = (long long)BB * HC * NN;
  conv1_bn_relu<<<dim3((unsigned)((hN + 255) / 256)), dim3(256), 0, stream>>>(xp, c1w, c1b, bn1g, bn1b, h1);
  conv2_bn_relu<<<dim3((unsigned)((hN + 255) / 256)), dim3(256), 0, stream>>>(h1, c2w, c2b, bn2g, bn2b, h2);
  sigma_k<<<dim3((BB * NN + 255) / 256), dim3(256), 0, stream>>>(h2, ow, ob, sigma);

  // 6) e = emb / (sigma + eps);  Gram G = e @ e^T  (WMMA)
  e_k<<<dim3((unsigned)((embN + 255) / 256)), dim3(256), 0, stream>>>(emb, sigma, e);
  launch_gemm(e, e, Wb, nullptr,
              NN, NN, DF, DF, DF, NN,
              (long long)NN * DF, 0, (long long)NN * DF, 0, (long long)NN * NN, 0,
              BB, 1, 1, 1.0f, stream);

  // 7) W = exp(-d2/(2*768)), d2 from Gram diagonal
  diag_k<<<dim3((BB * NN + 255) / 256), dim3(256), 0, stream>>>(Wb, diag);
  wexp_k<<<dim3((unsigned)((WN + 255) / 256)), dim3(256), 0, stream>>>(Wb, diag);

  // 8) top-k symmetric mask, apply
  zero_bytes<<<dim3(2048), dim3(256), 0, stream>>>((unsigned int*)mask, WN / 4);
  topk_mask<<<dim3((BB * NN + 255) / 256), dim3(256), 0, stream>>>(Wb, mask);
  apply_mask<<<dim3((unsigned)((WN + 255) / 256)), dim3(256), 0, stream>>>(Wb, mask);

  // 9) degree -> dsi, build augmented [A | I] with A = (1+eps)I - alpha*S
  rowsum_dsi<<<dim3(BB * NN), dim3(256), 0, stream>>>(Wb, dsi);
  build_aug<<<dim3((unsigned)((WN + 255) / 256)), dim3(256), 0, stream>>>(Wb, dsi, alp, aug);

  // 10) invert (Gauss-Jordan, partial pivot), one block per batch
  gauss_jordan<<<dim3(BB), dim3(1024), 0, stream>>>(aug);

  // 11) out = inv @ one_hot(labels): zero then column-scatter
  zero_bytes<<<dim3(8192), dim3(256), 0, stream>>>((unsigned int*)out, (long long)out_size);
  scatter_out<<<dim3(BB * NN), dim3(256), 0, stream>>>(aug, lab, out);
}
